// Model_56916906606870
// MI455X (gfx1250) — matrix operand order repaired and checked by reading, not compiled, not run on hardware
//
#include <hip/hip_runtime.h>
#include <hip/hip_bf16.h>
#include <math.h>
#include <stdint.h>

// ---------------------------------------------------------------------------
// All-tanh LSTM for MI455X (gfx1250, wave32, WMMA, async-to-LDS).
//   Phase 0a: x  (f32) -> Xb (bf16, packed dwords)            [128 MB ws]
//   Phase 0b: W  (f32) -> Wswz (bf16, B-fragment-swizzled)    [512 KB ws]
//   Phase 1 : xW = x @ W + b, bf16 WMMA GEMM; A tiles staged via
//             global_load_async_to_lds_b64 (double-buffered, s_wait_asynccnt);
//             stored t-major [T, B, 512] f32.                 [256 MB ws]
//   Phase 2 : persistent single-workgroup recurrence (32 waves, one WGP).
//             U swizzled into LDS (128 KB), h in LDS (bf16, 32 KB),
//             c entirely in registers; every WMMA operand = 2x b128 loads.
// ---------------------------------------------------------------------------

typedef __attribute__((ext_vector_type(16))) __bf16       v16bf;
typedef __attribute__((ext_vector_type(8)))  float        v8f;
typedef __attribute__((ext_vector_type(4)))  unsigned int u32x4;
typedef __attribute__((ext_vector_type(2)))  float        f32x2;

#define BATCH  128
#define SEQ    1024
#define INDIM  512
#define UNITS  128
#define GDIM   512                 // 4*UNITS
#define MROWS  (BATCH * SEQ)       // 131072

union BFrag {                      // one WMMA 16-bit operand = 8 VGPRs
    v16bf v;
    u32x4 q[2];
};

static __device__ __forceinline__ v8f wmma_bf16(v16bf a, v16bf b, v8f c) {
    return __builtin_amdgcn_wmma_f32_16x16x32_bf16(
        false, a, false, b, (short)0, c, false, false);
}

static __device__ __forceinline__ unsigned int pack_bf16(float lo, float hi) {
    unsigned short l = __builtin_bit_cast(unsigned short, (__bf16)lo);
    unsigned short h = __builtin_bit_cast(unsigned short, (__bf16)hi);
    return (unsigned int)l | ((unsigned int)h << 16);
}

// ---------------------------------------------------------------------------
// Phase 0a: x f32 -> bf16, two elements packed per dword (b64 in, b32 out)
// ---------------------------------------------------------------------------
__global__ void cvt_x_kernel(const float* __restrict__ x,
                             unsigned int* __restrict__ Xb, int ndw) {
    int i = blockIdx.x * blockDim.x + threadIdx.x;
    int stride = gridDim.x * blockDim.x;
    const f32x2* x2 = (const f32x2*)x;
    for (; i < ndw; i += stride) {
        f32x2 f = x2[i];
        Xb[i] = pack_bf16(f.x, f.y);
    }
}

// ---------------------------------------------------------------------------
// Phase 0b: W -> B-fragment-swizzled bf16.
// Layout: dword d = ((ntile*16 + kkIdx)*32 + lane)*8 + j
//   lane: n = lane&15 (column), half = lane>>4
//   dword j packs W[kk + half*16 + 2j][ncol], W[..+1][ncol]   (contiguous k order)
// => every fragment is 32 contiguous bytes per lane: 2x global_load_b128.
// ---------------------------------------------------------------------------
__global__ void cvt_w_swz_kernel(const float* __restrict__ W,
                                 unsigned int* __restrict__ Wswz) {
    int d = blockIdx.x * blockDim.x + threadIdx.x;     // 65536 dwords
    if (d >= 32 * 16 * 32 * 8) return;
    int j     = d & 7;
    int lane  = (d >> 3) & 31;
    int kkIdx = (d >> 8) & 15;
    int ntile = d >> 12;
    int ncol  = ntile * 16 + (lane & 15);
    int krow  = kkIdx * 32 + (lane >> 4) * 16 + 2 * j;
    Wswz[d] = pack_bf16(W[(size_t)krow * GDIM + ncol],
                        W[(size_t)(krow + 1) * GDIM + ncol]);
}

// ---------------------------------------------------------------------------
// Phase 1: xW[t][b][n] = sum_k x[b][t][k]*W[k][n] + bias[n]
// Block = 256 threads (8 waves): one 16-row M tile x 128 N cols.
// A tile (16x32 bf16, 1KB) async-staged to LDS, double buffered.
// ---------------------------------------------------------------------------
__global__ __launch_bounds__(256) void gemm_xw_kernel(
        const __bf16* __restrict__ Xb, const unsigned int* __restrict__ Wswz,
        const float* __restrict__ bias, float* __restrict__ xW) {
    __shared__ __align__(16) unsigned int As[2][256];   // 2 x 1KB A tiles

    const int mt   = blockIdx.x >> 2;            // M tile 0..8191
    const int ng   = blockIdx.x & 3;             // N group 0..3
    const int tid  = threadIdx.x;
    const int wave = tid >> 5;
    const int lane = tid & 31;
    const int n    = lane & 15;
    const int half = lane >> 4;
    const int ntile = ng * 8 + wave;
    const int ncol  = ntile * 16 + n;

    // async stagers: 128 threads, each moves 8 bytes of the 1KB tile
    const bool stager = tid < 128;
    const int  r = tid >> 3, c = tid & 7;        // row 0..15, 8B chunk 0..7
    const char* xrow = (const char*)Xb + (size_t)(mt * 16 + r) * (INDIM * 2) + c * 8;
    const unsigned int lds0 =
        (unsigned int)(uintptr_t)(void*)&As[0][0] + (unsigned int)(r * 64 + c * 8);
    const unsigned int lds1 =
        (unsigned int)(uintptr_t)(void*)&As[1][0] + (unsigned int)(r * 64 + c * 8);

    #define ISSUE_ASYNC(buf, kk)                                              \
        if (stager) {                                                         \
            unsigned int l = (buf) ? lds1 : lds0;                             \
            const char* g = xrow + (size_t)(kk) * 2;                          \
            asm volatile("global_load_async_to_lds_b64 %0, %1, off"           \
                         :: "v"(l), "v"(g) : "memory");                       \
        }

    ISSUE_ASYNC(0, 0);

    const u32x4* wfrag = (const u32x4*)Wswz + (size_t)ntile * 16 * 64 + lane * 2;
    v8f acc = {};
    for (int kkIdx = 0; kkIdx < 16; ++kkIdx) {
        if (kkIdx + 1 < 16) {
            ISSUE_ASYNC((kkIdx + 1) & 1, (kkIdx + 1) * 32);
            asm volatile("s_wait_asynccnt 0x1" ::: "memory");
        } else {
            asm volatile("s_wait_asynccnt 0x0" ::: "memory");
        }
        __syncthreads();                              // staged tile visible

        // A fragment: row n of the tile, 2x ds_load_b128 (contiguous per lane)
        const u32x4* arow = (const u32x4*)&As[kkIdx & 1][0] + n * 4;
        BFrag a, b;
        a.q[0] = arow[2 * half];
        a.q[1] = arow[2 * half + 1];
        // B fragment: 2x global_load_b128 from L2-resident swizzled W
        const u32x4* p = wfrag + kkIdx * 64;
        b.q[0] = p[0];
        b.q[1] = p[1];
        acc = wmma_bf16(a.v, b.v, acc);
        __syncthreads();                              // before buffer reuse
    }
    #undef ISSUE_ASYNC

    const float bv = bias[ncol];
    #pragma unroll
    for (int j = 0; j < 8; ++j) {
        int r2 = mt * 16 + j + 8 * half;             // r = b*SEQ + t
        int t  = r2 & (SEQ - 1);
        int bi = r2 >> 10;
        xW[((size_t)t * BATCH + bi) * GDIM + ncol] = acc[j] + bv;
    }
}

// ---------------------------------------------------------------------------
// Phase 2: sequential LSTM, one workgroup (1024 threads = 32 waves, one WGP).
// Wave w: mi = w>>2 (batch tile), jg = w&3 (unit-column group); wave owns all
// four gate tiles for its elements -> c update is register-only.
// ---------------------------------------------------------------------------
__global__ __launch_bounds__(1024) void lstm_seq_kernel(
        const float* __restrict__ U, const float* __restrict__ xW,
        float* __restrict__ out) {
    __shared__ __align__(16) unsigned int Uswz[32 * 4 * 32 * 8];  // 128 KB
    __shared__ __align__(16) __bf16       hS[BATCH * UNITS];      // 32 KB

    const int tid  = threadIdx.x;
    const int lane = tid & 31;
    const int wave = tid >> 5;
    const int mi   = wave >> 2;
    const int jg   = wave & 3;
    const int n    = lane & 15;
    const int half = lane >> 4;

    // stage U into B-fragment-swizzled LDS layout (fragments: 4 kk x 32 ntile)
    for (int i = tid; i < 32 * 4 * 32 * 8; i += 1024) {
        int j  = i & 7;
        int ln = (i >> 3) & 31;
        int kkIdx = (i >> 8) & 3;
        int nt = i >> 10;
        int nc = nt * 16 + (ln & 15);
        int kr = kkIdx * 32 + (ln >> 4) * 16 + 2 * j;
        Uswz[i] = pack_bf16(U[(size_t)kr * GDIM + nc],
                            U[(size_t)(kr + 1) * GDIM + nc]);
    }
    for (int i = tid; i < BATCH * UNITS; i += 1024) hS[i] = (__bf16)0.0f;
    __syncthreads();

    const u32x4* U4 = (const u32x4*)Uswz;
    const u32x4* h4base = (const u32x4*)hS + (size_t)(mi * 16 + n) * 16; // row = 16 u32x4

    float creg[2][8], hreg[2][8];
    #pragma unroll
    for (int s = 0; s < 2; ++s)
        #pragma unroll
        for (int j = 0; j < 8; ++j) { creg[s][j] = 0.0f; hreg[s][j] = 0.0f; }

    for (int t = 0; t < SEQ; ++t) {
        v8f acc[4][2];
        #pragma unroll
        for (int g = 0; g < 4; ++g) { v8f z = {}; acc[g][0] = z; acc[g][1] = z; }

        if (t + 1 < SEQ)    // pull next step's xW slab toward the caches
            __builtin_prefetch(xW + ((size_t)(t + 1) * BATCH + mi * 16) * GDIM, 0, 0);

        // z = h @ U : 4 K-steps of 32; every operand is 2x ds_load_b128
        #pragma unroll
        for (int kkIdx = 0; kkIdx < 4; ++kkIdx) {
            BFrag a;
            a.q[0] = h4base[kkIdx * 4 + 2 * half];
            a.q[1] = h4base[kkIdx * 4 + 2 * half + 1];
            #pragma unroll
            for (int g = 0; g < 4; ++g) {
                #pragma unroll
                for (int s = 0; s < 2; ++s) {
                    int f = (g * 8 + jg * 2 + s) * 4 + kkIdx;   // fragment id
                    const u32x4* p = U4 + (size_t)f * 64 + lane * 2;
                    BFrag b;
                    b.q[0] = p[0];
                    b.q[1] = p[1];
                    acc[g][s] = wmma_bf16(a.v, b.v, acc[g][s]);
                }
            }
        }
        __syncthreads();                 // all hS reads for step t done

        // gates + state update (register-only c), h' -> LDS bf16
        #pragma unroll
        for (int s = 0; s < 2; ++s) {
            int ucol = (jg * 2 + s) * 16 + n;
            #pragma unroll
            for (int j = 0; j < 8; ++j) {
                int brow = mi * 16 + j + 8 * half;
                const float* zr = xW + ((size_t)t * BATCH + brow) * GDIM;
                float zi = acc[0][s][j] + zr[0 * UNITS + ucol];
                float zf = acc[1][s][j] + zr[1 * UNITS + ucol];
                float zg = acc[2][s][j] + zr[2 * UNITS + ucol];
                float zo = acc[3][s][j] + zr[3 * UNITS + ucol];
                float cn = tanhf(zf) * creg[s][j] + tanhf(zi) * tanhf(zg);
                float hn = tanhf(zo) * tanhf(cn);
                creg[s][j] = cn;
                hreg[s][j] = hn;
                hS[brow * UNITS + ucol] = (__bf16)hn;
            }
        }
        __syncthreads();                 // h' visible before next step
    }

    #pragma unroll
    for (int s = 0; s < 2; ++s) {
        int ucol = (jg * 2 + s) * 16 + n;
        #pragma unroll
        for (int j = 0; j < 8; ++j) {
            int brow = mi * 16 + j + 8 * half;
            out[brow * UNITS + ucol] = hreg[s][j];
        }
    }
}

// ---------------------------------------------------------------------------
extern "C" void kernel_launch(void* const* d_in, const int* in_sizes, int n_in,
                              void* d_out, int out_size, void* d_ws, size_t ws_size,
                              hipStream_t stream) {
    const float* x = (const float*)d_in[0];   // [B, T, 512]
    const float* W = (const float*)d_in[1];   // [512, 512]
    const float* U = (const float*)d_in[2];   // [128, 512]
    const float* b = (const float*)d_in[3];   // [512]
    float* out = (float*)d_out;               // [128, 128]

    // ws: | xW f32 [T,B,512] 256MB | Xb bf16 128MB | Wswz 512KB |
    float*        xW   = (float*)d_ws;
    char*         base = (char*)d_ws;
    __bf16*       Xb   = (__bf16*)(base + (size_t)MROWS * GDIM * sizeof(float));
    unsigned int* Wswz = (unsigned int*)(base + (size_t)MROWS * GDIM * sizeof(float)
                                              + (size_t)MROWS * INDIM * sizeof(__bf16));

    cvt_x_kernel<<<dim3(8192), dim3(256), 0, stream>>>(x, (unsigned int*)Xb,
                                                       MROWS * INDIM / 2);
    cvt_w_swz_kernel<<<dim3(256), dim3(256), 0, stream>>>(W, Wswz);
    gemm_xw_kernel<<<dim3((MROWS / 16) * 4), dim3(256), 0, stream>>>(Xb, Wswz, b, xW);
    lstm_seq_kernel<<<dim3(1), dim3(1024), 0, stream>>>(U, xW, out);
}
